// NeighAttnAgg_16398185136399
// MI455X (gfx1250) — matrix operand order repaired and checked by reading, not compiled, not run on hardware
//
#include <hip/hip_runtime.h>
#include <hip/hip_bf16.h>
#include <stdint.h>

#define N_HEADS 6
#define KWIN    7
#define DILA    2
#define CH      192
#define HDIMV   128
#define WDIMV   128
#define BDIMV   8
#define NPIX    (BDIMV*HDIMV*WDIMV)   /* 131072 */
#define TILE    16
#define NW      36864                 /* 192*192 */

typedef __attribute__((ext_vector_type(16))) __bf16   v16bf;
typedef __attribute__((ext_vector_type(8)))  float    v8f;
typedef __attribute__((ext_vector_type(4)))  unsigned u32x4;
typedef __attribute__((ext_vector_type(4)))  float    f32x4;

union V16U { v16bf v; unsigned short u[16]; u32x4 q[2]; };
union V8FU { v8f v; float f[8]; };

__device__ __forceinline__ unsigned short f2bf(float f) {
  unsigned u = __float_as_uint(f);
  unsigned r = u + 0x7FFFu + ((u >> 16) & 1u);   // round to nearest even
  return (unsigned short)(r >> 16);
}

// ---------------- kernel 0: weights fp32 -> bf16 ----------------
__global__ void convert_w(const float* __restrict__ wv, const float* __restrict__ wp,
                          unsigned short* __restrict__ wv_bf,
                          unsigned short* __restrict__ wp_bf) {
  int i = blockIdx.x * blockDim.x + threadIdx.x;
  if (i < NW)            wv_bf[i]      = f2bf(wv[i]);
  else if (i < 2 * NW)   wp_bf[i - NW] = f2bf(wp[i - NW]);
}

// ---------------- kernel 1: v = x @ Wv + bv (bf16 WMMA, f32 acc) ----------------
__global__ __launch_bounds__(384) void v_gemm(const float* __restrict__ x,
                                              const unsigned short* __restrict__ wv_bf,
                                              const float* __restrict__ bv,
                                              float* __restrict__ v_out) {
  __shared__ unsigned short lxa[TILE * CH];     // 16x192 bf16 A tile (6KB)
  const int pix0 = blockIdx.x * TILE;

  for (int t = threadIdx.x; t < TILE * CH; t += blockDim.x)
    lxa[t] = f2bf(x[(size_t)pix0 * CH + t]);
  __syncthreads();

  const int wave    = threadIdx.x >> 5;         // 0..11 -> N tile
  const int lane    = threadIdx.x & 31;
  const int laneoff = (lane >> 4) << 3;         // 0 or 8 (A K sub-block)
  const int m       = lane & 15;

  v8f acc = {};
  #pragma unroll
  for (int ks = 0; ks < CH; ks += 32) {
    V16U a, b;
    // A fragment: row m, K = ks+laneoff+[0..7] and ks+laneoff+16+[0..7]
    const u32x4* pa = (const u32x4*)&lxa[m * CH + ks + laneoff];
    a.q[0] = pa[0];
    a.q[1] = pa[2];                             // +16 bf16 elements
    // B fragment: column n = wave*16+m, K = ks+laneoff+[0..7] and ks+laneoff+16+[0..7]
    for (int s2 = 0; s2 < 16; ++s2) b.u[s2] = wv_bf[(size_t)(ks + laneoff + (s2 & 7) + ((s2 >> 3) << 4)) * CH + wave * 16 + m];
    acc = __builtin_amdgcn_wmma_f32_16x16x32_bf16(false, a.v, false, b.v,
                                                  (short)0, acc, false, false);
  }

  const int n     = wave * 16 + m;
  const float bia = bv[n];
  const int mrow0 = (lane >> 4) << 3;
  V8FU r; r.v = acc;
  #pragma unroll
  for (int rr = 0; rr < 8; ++rr)
    v_out[(size_t)(pix0 + mrow0 + rr) * CH + n] = r.f[rr] + bia;
}

// -------- kernel 2: fused softmax + neighborhood aggregation + Wp projection --------
__global__ __launch_bounds__(384) void agg_proj(const float* __restrict__ attn,
                                                const float* __restrict__ v,
                                                const unsigned short* __restrict__ wp_bf,
                                                const float* __restrict__ bp,
                                                float* __restrict__ out) {
  __shared__ unsigned short lagg[TILE * CH];    // 16x192 bf16 aggregated tile

  // block -> (b, row i, 16-column tile j0); grid = 8*128*8 = 8192
  const int jt = blockIdx.x & 7;
  const int i  = (blockIdx.x >> 3) & 127;
  const int b  = blockIdx.x >> 10;
  const int j0 = jt * 16;

  // thread -> (pixel in tile, 8-channel slice). 16 px * 24 slices = 384 threads
  const int pix  = threadIdx.x & 15;
  const int sub  = threadIdx.x >> 4;            // 0..23
  const int c0   = sub * 8;                     // channel octet (within one head: 8 | 32)
  const int head = c0 >> 5;
  const int j    = j0 + pix;

  // NATTEN dilated, border-clamped neighbor coordinates (d=2 -> Li=64, Li-K=57)
  int ri[KWIN], cj[KWIN];
  {
    int g = i & 1, ii = i >> 1;
    int st = ii - 3; st = st < 0 ? 0 : (st > 57 ? 57 : st);
    #pragma unroll
    for (int p = 0; p < KWIN; ++p) ri[p] = (st + p) * DILA + g;
  }
  {
    int g = j & 1, jj = j >> 1;
    int st = jj - 3; st = st < 0 ? 0 : (st > 57 ? 57 : st);
    #pragma unroll
    for (int q = 0; q < KWIN; ++q) cj[q] = (st + q) * DILA + g;
  }

  // in-register softmax over the 49 logits
  const float* ap = attn + ((((size_t)b * N_HEADS + head) * HDIMV + i) * WDIMV + j) * 49;
  float ev[49], mx = -3.4e38f;
  #pragma unroll
  for (int t = 0; t < 49; ++t) { ev[t] = ap[t]; mx = fmaxf(mx, ev[t]); }
  float s = 0.f;
  #pragma unroll
  for (int t = 0; t < 49; ++t) { ev[t] = __expf(ev[t] - mx); s += ev[t]; }
  const float inv = 1.0f / s;

  // weighted aggregation over 7x7 neighborhood (v is L2-resident)
  float acc[8] = {0.f, 0.f, 0.f, 0.f, 0.f, 0.f, 0.f, 0.f};
  #pragma unroll
  for (int p = 0; p < KWIN; ++p) {
    const size_t rowbase = ((size_t)b * HDIMV + ri[p]) * WDIMV;
    #pragma unroll
    for (int q = 0; q < KWIN; ++q) {
      const float w = ev[p * 7 + q] * inv;
      const f32x4* vp = (const f32x4*)&v[(rowbase + cj[q]) * CH + c0];
      f32x4 v0 = vp[0], v1 = vp[1];
      acc[0] += w * v0.x; acc[1] += w * v0.y; acc[2] += w * v0.z; acc[3] += w * v0.w;
      acc[4] += w * v1.x; acc[5] += w * v1.y; acc[6] += w * v1.z; acc[7] += w * v1.w;
    }
  }
  #pragma unroll
  for (int k = 0; k < 8; ++k) lagg[pix * CH + c0 + k] = f2bf(acc[k]);
  __syncthreads();

  // projection: out_tile = agg_tile @ Wp + bp  (same WMMA scheme as v_gemm)
  const int wave    = threadIdx.x >> 5;
  const int lane    = threadIdx.x & 31;
  const int laneoff = (lane >> 4) << 3;
  const int m       = lane & 15;

  v8f cacc = {};
  #pragma unroll
  for (int ks = 0; ks < CH; ks += 32) {
    V16U a, bfr;
    const u32x4* pa = (const u32x4*)&lagg[m * CH + ks + laneoff];
    a.q[0] = pa[0];
    a.q[1] = pa[2];
    for (int s2 = 0; s2 < 16; ++s2) bfr.u[s2] = wp_bf[(size_t)(ks + laneoff + (s2 & 7) + ((s2 >> 3) << 4)) * CH + wave * 16 + m];
    cacc = __builtin_amdgcn_wmma_f32_16x16x32_bf16(false, a.v, false, bfr.v,
                                                   (short)0, cacc, false, false);
  }

  const int n      = wave * 16 + m;
  const float bia  = bp[n];
  const int mrow0  = (lane >> 4) << 3;
  const size_t ob  = ((size_t)b * HDIMV + i) * WDIMV + j0;
  V8FU r; r.v = cacc;
  #pragma unroll
  for (int rr = 0; rr < 8; ++rr)
    out[(ob + mrow0 + rr) * CH + n] = r.f[rr] + bia;
}

extern "C" void kernel_launch(void* const* d_in, const int* in_sizes, int n_in,
                              void* d_out, int out_size, void* d_ws, size_t ws_size,
                              hipStream_t stream) {
  const float* x    = (const float*)d_in[0];
  const float* attn = (const float*)d_in[1];
  const float* Wv   = (const float*)d_in[2];
  const float* bv   = (const float*)d_in[3];
  const float* Wp   = (const float*)d_in[4];
  const float* bp   = (const float*)d_in[5];
  float* out        = (float*)d_out;

  // workspace layout (256B-aligned sections):
  //   [0,       73728)  Wv bf16
  //   [73728,  147456)  Wp bf16
  //   [147456, +96MiB)  v fp32 [131072][192]
  unsigned short* wv_bf = (unsigned short*)d_ws;
  unsigned short* wp_bf = (unsigned short*)((char*)d_ws + 73728);
  float*          vbuf  = (float*)((char*)d_ws + 147456);

  convert_w<<<dim3((2 * NW) / 256), dim3(256), 0, stream>>>(Wv, Wp, wv_bf, wp_bf);
  v_gemm<<<dim3(NPIX / TILE), dim3(384), 0, stream>>>(x, wv_bf, bv, vbuf);
  agg_proj<<<dim3(NPIX / TILE), dim3(384), 0, stream>>>(attn, vbuf, wp_bf, bp, out);
}
